// MixerGatedDeltaNet_10806137717337
// MI455X (gfx1250) — matrix operand order repaired and checked by reading, not compiled, not run on hardware
//
#include <hip/hip_runtime.h>
#include <hip/hip_bf16.h>
#include <math.h>

// ---------------- problem constants ----------------
#define B_      2
#define L_      1024
#define D_      768
#define H_      12
#define DK_     64
#define DV_     128
#define KEY_DIM 768
#define VAL_DIM 1536
#define QKV_COLS 3072          // 2*KEY_DIM + VAL_DIM (WMMA-covered part of IN_DIM)
#define IN_DIM  3096
#define M_TOK   (B_*L_)        // 2048
#define KC_     4
#define SCALE_  0.125f         // DK^-0.5
#define EPS_NORM 1e-5f
#define EPS_L2   1e-6f

typedef __attribute__((ext_vector_type(16))) __bf16       v16bf;
typedef __attribute__((ext_vector_type(8)))  float        v8f;
typedef __attribute__((ext_vector_type(8)))  unsigned int v8u;

__device__ __forceinline__ unsigned short f32_to_bf16(float f) {
  unsigned int u = __builtin_bit_cast(unsigned int, f);
  u += 0x7FFFu + ((u >> 16) & 1u);     // round-to-nearest-even
  return (unsigned short)(u >> 16);
}
__device__ __forceinline__ float sigmoidf_(float x) { return 1.f / (1.f + __expf(-x)); }

// load one 16-element bf16 WMMA fragment (two aligned 16B chunks) from a K-contiguous row
__device__ __forceinline__ v16bf load_frag(const unsigned short* rowp, int o0, int o1) {
  uint4 c0 = *(const uint4*)(rowp + o0);
  uint4 c1 = *(const uint4*)(rowp + o1);
  v8u u = {c0.x, c0.y, c0.z, c0.w, c1.x, c1.y, c1.z, c1.w};
  return __builtin_bit_cast(v16bf, u);
}

// ---------------- pack kernels ----------------
__global__ void pack_bf16(const float* __restrict__ in, unsigned short* __restrict__ out, int n) {
  int i = blockIdx.x * blockDim.x + threadIdx.x;
  if (i < n) out[i] = f32_to_bf16(in[i]);
}

// out[n*rows + k] = bf16(in[k*cols_total + n]);  n < cols_out  (transpose to K-contiguous)
__global__ void pack_bf16_T(const float* __restrict__ in, unsigned short* __restrict__ out,
                            int rows, int cols_total, int cols_out) {
  int i = blockIdx.x * blockDim.x + threadIdx.x;
  if (i >= rows * cols_out) return;
  int n = i / rows, k = i % rows;
  out[(size_t)n * rows + k] = f32_to_bf16(in[(size_t)k * cols_total + n]);
}

// ---------------- WMMA bf16 GEMM: C[M,N] = A[M,K] * B[K,N], BT is [N,K] ----------------
// 256 threads = 8 waves arranged 2(M) x 4(N). Each wave: 32x32 C block (2x2 WMMA tiles)
// -> workgroup block tile 64x128. 8 b128 loads : 4 wmma per K-step (16 FLOP/byte).
__global__ void gemm_bf16_wmma(const unsigned short* __restrict__ A,
                               const unsigned short* __restrict__ BT,
                               float* __restrict__ C,
                               int M, int N, int K, int ldc) {
  int wave = threadIdx.x >> 5;
  int lane = threadIdx.x & 31;
  int wm = wave >> 2, wn = wave & 3;
  int m0 = blockIdx.y * 64  + wm * 32;
  int n0 = blockIdx.x * 128 + wn * 32;
  int lh = lane & 15;
  // A (16x32 bf16): lanes 0-15 hold K=[0,16), lanes 16-31 K=[16,32), per-lane K-contiguous
  int aoff = (lane < 16) ? 0 : 16;
  // B (32x16 bf16): lanes 0-15 hold K=[0,16), lanes 16-31 K=[16,32), per-lane K-contiguous
  int boff = (lane < 16) ? 0 : 16;

  const unsigned short* a0row = A  + (size_t)(m0 + lh)      * K;
  const unsigned short* a1row = A  + (size_t)(m0 + 16 + lh) * K;
  const unsigned short* b0row = BT + (size_t)(n0 + lh)      * K;
  const unsigned short* b1row = BT + (size_t)(n0 + 16 + lh) * K;

  v8f c00 = {}, c01 = {}, c10 = {}, c11 = {};
  for (int kt = 0; kt < K; kt += 32) {
    // prefetch next K-chunk (gfx1250 global_prefetch_b8); workspace regions are
    // adjacent so the final +64 elements stay inside d_ws.
    __builtin_prefetch(a0row + kt + 64, 0, 0);
    __builtin_prefetch(b0row + kt + 64, 0, 0);
    v16bf a0 = load_frag(a0row, kt + aoff, kt + aoff + 8);
    v16bf a1 = load_frag(a1row, kt + aoff, kt + aoff + 8);
    v16bf b0 = load_frag(b0row, kt + boff, kt + boff + 8);
    v16bf b1 = load_frag(b1row, kt + boff, kt + boff + 8);
    c00 = __builtin_amdgcn_wmma_f32_16x16x32_bf16(false, a0, false, b0, (short)0, c00, false, false);
    c01 = __builtin_amdgcn_wmma_f32_16x16x32_bf16(false, a0, false, b1, (short)0, c01, false, false);
    c10 = __builtin_amdgcn_wmma_f32_16x16x32_bf16(false, a1, false, b0, (short)0, c10, false, false);
    c11 = __builtin_amdgcn_wmma_f32_16x16x32_bf16(false, a1, false, b1, (short)0, c11, false, false);
  }
  // C/D layout: VGPR r -> M = r (lanes 0-15) / 8+r (lanes 16-31), N = lane&15
  int mb = m0 + ((lane >> 4) << 3);
  #pragma unroll
  for (int r = 0; r < 8; ++r) {
    C[(size_t)(mb + r)      * ldc + n0 + lh]      = c00[r];
    C[(size_t)(mb + r)      * ldc + n0 + 16 + lh] = c01[r];
    C[(size_t)(mb + 16 + r) * ldc + n0 + lh]      = c10[r];
    C[(size_t)(mb + 16 + r) * ldc + n0 + 16 + lh] = c11[r];
  }
}

// ---------------- tail: b/a columns of W_in -> beta, g ----------------
__global__ void tail_ba(const float* __restrict__ hs, const float* __restrict__ W_in,
                        const float* __restrict__ dt_bias, const float* __restrict__ A_log,
                        float* __restrict__ beta, float* __restrict__ g) {
  int idx = blockIdx.x * blockDim.x + threadIdx.x;
  if (idx >= M_TOK * H_) return;
  int m = idx / H_, h = idx % H_;
  const float* x = hs + (size_t)m * D_;
  float accb = 0.f, acca = 0.f;
  for (int k = 0; k < D_; ++k) {
    float xv = x[k];
    accb += xv * W_in[(size_t)k * IN_DIM + QKV_COLS + h];
    acca += xv * W_in[(size_t)k * IN_DIM + QKV_COLS + H_ + h];
  }
  beta[idx] = sigmoidf_(accb);
  float ap = acca + dt_bias[h];
  float sp = (ap > 20.f) ? ap : log1pf(__expf(ap));   // softplus
  g[idx] = -__expf(A_log[h]) * sp;
}

// ---------------- short conv + SiLU + L2 norm for q,k (one WG per (b,l,h)) ----------------
__global__ void conv_qk(const float* __restrict__ qkv, const float* __restrict__ qw,
                        const float* __restrict__ kw, float* __restrict__ q,
                        float* __restrict__ k) {
  int m = blockIdx.x / H_;
  int h = blockIdx.x % H_;
  int l = m % L_;
  int tid = threadIdx.x;            // 0..63
  int c = h * DK_ + tid;
  float accq = 0.f, acck = 0.f;
  #pragma unroll
  for (int j = 0; j < KC_; ++j) {
    int li = l - (KC_ - 1) + j;
    if (li >= 0) {
      const float* row = qkv + (size_t)(m - (KC_ - 1) + j) * QKV_COLS;
      accq += row[c] * qw[c * KC_ + j];
      acck += row[KEY_DIM + c] * kw[c * KC_ + j];
    }
  }
  accq *= sigmoidf_(accq);          // SiLU
  acck *= sigmoidf_(acck);
  __shared__ float sq[DK_], sk[DK_];
  sq[tid] = accq * accq; sk[tid] = acck * acck;
  __syncthreads();
  for (int s = 32; s > 0; s >>= 1) {
    if (tid < s) { sq[tid] += sq[tid + s]; sk[tid] += sk[tid + s]; }
    __syncthreads();
  }
  q[(size_t)m * KEY_DIM + c] = accq * rsqrtf(sq[0] + EPS_L2);
  k[(size_t)m * KEY_DIM + c] = acck * rsqrtf(sk[0] + EPS_L2);
}

// ---------------- short conv + SiLU for v ----------------
__global__ void conv_v(const float* __restrict__ qkv, const float* __restrict__ vw,
                       float* __restrict__ v) {
  int m = blockIdx.x;
  int l = m % L_;
  for (int c = threadIdx.x; c < VAL_DIM; c += blockDim.x) {
    float acc = 0.f;
    #pragma unroll
    for (int j = 0; j < KC_; ++j) {
      int li = l - (KC_ - 1) + j;
      if (li >= 0)
        acc += qkv[(size_t)(m - (KC_ - 1) + j) * QKV_COLS + 2 * KEY_DIM + c] * vw[c * KC_ + j];
    }
    v[(size_t)m * VAL_DIM + c] = acc * sigmoidf_(acc);
  }
}

// ---------------- gated delta rule scan: 1 WG per (b,h), 2 lanes per DV column ----------------
__global__ void delta_scan(const float* __restrict__ q, const float* __restrict__ k,
                           const float* __restrict__ v, const float* __restrict__ g,
                           const float* __restrict__ beta, float* __restrict__ o) {
  int b = blockIdx.x / H_;
  int h = blockIdx.x % H_;
  int tid = threadIdx.x;            // 0..255
  int c   = tid >> 1;               // dv column 0..127
  int sub = tid & 1;                // dk half
  int kb  = sub * 32;
  float S[32];
  #pragma unroll
  for (int i = 0; i < 32; ++i) S[i] = 0.f;
  __shared__ float shk[DK_], shq[DK_], shv[DV_];
  __shared__ float shg[2];
  for (int t = 0; t < L_; ++t) {
    size_t m = (size_t)b * L_ + t;
    __syncthreads();
    if (tid < DK_)            shk[tid]           = k[m * KEY_DIM + h * DK_ + tid];
    else if (tid < 2 * DK_)   shq[tid - DK_]     = q[m * KEY_DIM + h * DK_ + (tid - DK_)];
    else                      shv[tid - 2 * DK_] = v[m * VAL_DIM + h * DV_ + (tid - 2 * DK_)];
    if (tid == 0) { shg[0] = __expf(g[m * H_ + h]); shg[1] = beta[m * H_ + h]; }
    __syncthreads();
    float decay = shg[0], bta = shg[1];
    float p = 0.f;
    #pragma unroll
    for (int i = 0; i < 32; ++i) p += shk[kb + i] * S[i];
    p *= decay;
    float ks  = p + __shfl_xor(p, 1, 32);
    float err = shv[c] - ks;
    float po = 0.f;
    #pragma unroll
    for (int i = 0; i < 32; ++i) {
      float kv = shk[kb + i];
      float s  = decay * S[i] + bta * kv * err;
      S[i] = s;
      po += shq[kb + i] * s;
    }
    float ov = (po + __shfl_xor(po, 1, 32)) * SCALE_;
    if (sub == 0) o[m * VAL_DIM + h * DV_ + c] = ov;
  }
}

// ---------------- output RMS-norm * w * silu-gate (in place on d_out) ----------------
__global__ void out_norm_gate(float* __restrict__ o, const float* __restrict__ gate,
                              const float* __restrict__ w) {
  int m = blockIdx.x / H_;
  int h = blockIdx.x % H_;
  int c = threadIdx.x;              // 0..127
  size_t idx = (size_t)m * VAL_DIM + h * DV_ + c;
  float ov = o[idx];
  __shared__ float red[DV_];
  red[c] = ov * ov;
  __syncthreads();
  for (int s = 64; s > 0; s >>= 1) { if (c < s) red[c] += red[c + s]; __syncthreads(); }
  float var = red[0] * (1.f / DV_);
  float gv  = gate[idx];
  o[idx] = ov * rsqrtf(var + EPS_NORM) * w[c] * gv * sigmoidf_(gv);
}

// ---------------- launch ----------------
extern "C" void kernel_launch(void* const* d_in, const int* in_sizes, int n_in,
                              void* d_out, int out_size, void* d_ws, size_t ws_size,
                              hipStream_t stream) {
  const float* hs    = (const float*)d_in[0];
  const float* W_in  = (const float*)d_in[1];
  const float* W_g   = (const float*)d_in[2];
  const float* dt_b  = (const float*)d_in[3];
  const float* A_log = (const float*)d_in[4];
  const float* qw    = (const float*)d_in[5];
  const float* kw    = (const float*)d_in[6];
  const float* vw    = (const float*)d_in[7];
  const float* onw   = (const float*)d_in[8];
  float* out = (float*)d_out;

  char* ws = (char*)d_ws;
  auto alloc = [&](size_t bytes) {
    char* p = ws;
    ws += (bytes + 255) & ~(size_t)255;
    return p;
  };
  unsigned short* hs_bf   = (unsigned short*)alloc((size_t)M_TOK * D_ * 2);
  unsigned short* winT_bf = (unsigned short*)alloc((size_t)QKV_COLS * D_ * 2);
  unsigned short* wgT_bf  = (unsigned short*)alloc((size_t)VAL_DIM * D_ * 2);
  float* qkv_f  = (float*)alloc((size_t)M_TOK * QKV_COLS * 4);
  float* gate_f = (float*)alloc((size_t)M_TOK * VAL_DIM * 4);
  float* q_f    = (float*)alloc((size_t)M_TOK * KEY_DIM * 4);
  float* k_f    = (float*)alloc((size_t)M_TOK * KEY_DIM * 4);
  float* v_f    = (float*)alloc((size_t)M_TOK * VAL_DIM * 4);
  float* beta_f = (float*)alloc((size_t)M_TOK * H_ * 4);
  float* g_f    = (float*)alloc((size_t)M_TOK * H_ * 4);

  // 1) pack operands to bf16 (A row-major K-contig; B transposed -> K-contig rows)
  {
    int n = M_TOK * D_;
    pack_bf16<<<(n + 255) / 256, 256, 0, stream>>>(hs, hs_bf, n);
    int nw = QKV_COLS * D_;
    pack_bf16_T<<<(nw + 255) / 256, 256, 0, stream>>>(W_in, winT_bf, D_, IN_DIM, QKV_COLS);
    int ng = VAL_DIM * D_;
    pack_bf16_T<<<(ng + 255) / 256, 256, 0, stream>>>(W_g, wgT_bf, D_, VAL_DIM, VAL_DIM);
  }
  // 2) WMMA GEMMs (block tile 64x128, wave tile 32x32)
  gemm_bf16_wmma<<<dim3(QKV_COLS / 128, M_TOK / 64), 256, 0, stream>>>(
      hs_bf, winT_bf, qkv_f, M_TOK, QKV_COLS, D_, QKV_COLS);
  gemm_bf16_wmma<<<dim3(VAL_DIM / 128, M_TOK / 64), 256, 0, stream>>>(
      hs_bf, wgT_bf, gate_f, M_TOK, VAL_DIM, D_, VAL_DIM);
  // 3) b/a tail -> beta, g
  tail_ba<<<(M_TOK * H_ + 255) / 256, 256, 0, stream>>>(hs, W_in, dt_b, A_log, beta_f, g_f);
  // 4) short convs
  conv_qk<<<M_TOK * H_, DK_, 0, stream>>>(qkv_f, qw, kw, q_f, k_f);
  conv_v<<<M_TOK, 256, 0, stream>>>(qkv_f, vw, v_f);
  // 5) sequential gated delta rule -> pre-norm o into d_out
  delta_scan<<<B_ * H_, 256, 0, stream>>>(q_f, k_f, v_f, g_f, beta_f, out);
  // 6) RMS-norm + gated SiLU, in place
  out_norm_gate<<<M_TOK * H_, DV_, 0, stream>>>(out, gate_f, onw);
}
